// MoEViT_85736137163327
// MI455X (gfx1250) — matrix operand order repaired and checked by reading, not compiled, not run on hardware
//
#include <hip/hip_runtime.h>
#include <hip/hip_bf16.h>
#include <cstdint>
#include <cstddef>

// ---------------- model dims ----------------
#define BB      8
#define DD      768
#define NHH     12
#define HDD     64
#define EE      4
#define HIDD    3072
#define DEPTHH  12
#define NCC     1000
#define SS      197          // real sequence length (196 patches + cls)
#define SP      256          // padded sequence length (4 n-tiles of 64)
#define MPB     (BB*SP)      // 2048 padded token rows
#define NPATCH  196
#define MPAT    (BB*NPATCH)  // 1568 patch rows (98 tiles of 16 exactly)

typedef __bf16 bf16_t;
typedef __attribute__((ext_vector_type(16))) __bf16 v16bf;
typedef __attribute__((ext_vector_type(8)))  __bf16 v8bf;
typedef __attribute__((ext_vector_type(8)))  float  v8f;

#define LDS_PITCH 40  // bf16 elems per staged B row: 80B pitch, 16B aligned, bank-spread

#if __has_builtin(__builtin_amdgcn_global_load_async_to_lds_b128) && \
    __has_builtin(__builtin_amdgcn_s_wait_asynccnt)
#define MOE_ASYNC 1
// builtin signature: (int4 addrspace(1)* src, int4 addrspace(3)* dst, imm offset, imm cpol)
typedef int a4i __attribute__((vector_size(16)));
typedef a4i __attribute__((address_space(1)))* a4i_g;
typedef a4i __attribute__((address_space(3)))* a4i_l;
#else
#define MOE_ASYNC 0
#endif

__device__ __forceinline__ v16bf frag_pair(const bf16_t* p0, const bf16_t* p1)
{
  union { v16bf v; v8bf h[2]; } u;
  u.h[0] = *(const v8bf*)p0;
  u.h[1] = *(const v8bf*)p1;
  return u.v;
}

__device__ __forceinline__ void acc_zero(v8f acc[4])
{
#pragma unroll
  for (int t = 0; t < 4; ++t)
#pragma unroll
    for (int r = 0; r < 8; ++r) acc[t][r] = 0.0f;
}

__device__ __forceinline__ float gelu_tanh(float x)
{
  const float c = 0.7978845608028654f;  // sqrt(2/pi)
  float u = c * (x + 0.044715f * x * x * x);
  return 0.5f * x * (1.0f + tanhf(u));
}

// -------------------------------------------------------------------------
// WMMA GEMM core: block = 128 threads = 4 waves, all call together.
// Each wave computes a 16(M) x 64(N) f32 tile: C = A(MxK) * B(NxK)^T.
//   arow : per-lane pointer to A row (row = m0 + (lane&15)), bf16, contiguous K
//   Bm   : [N x K] row-major bf16, block covers columns n0..n0+63
// B tiles (64x32) are staged in LDS cooperatively (async global->LDS when
// available), then read back as WMMA B fragments via ds_load_b128.
// Fragment layouts per CDNA5 ISA 7.12.2 (wave32).
// -------------------------------------------------------------------------
__device__ __forceinline__ void gemm_core(const bf16_t* arow,
                                          const bf16_t* __restrict__ Bm, long ldb,
                                          int n0, int K, bf16_t* lds, v8f acc[4])
{
  const int tid  = threadIdx.x;
  const int lane = tid & 31;
  const int half = lane >> 4;
  const int lr   = lane & 15;
  for (int k0 = 0; k0 < K; k0 += 32) {
    __syncthreads();  // protect LDS reads of previous iteration
#pragma unroll
    for (int pass = 0; pass < 2; ++pass) {
      int c   = tid + pass * 128;     // 256 chunks of 8 bf16 = 64 rows x 32 cols
      int row = c >> 2;
      int col = (c & 3) << 3;
      const bf16_t* g = Bm + (long)(n0 + row) * ldb + (k0 + col);
      bf16_t* l = lds + row * LDS_PITCH + col;
#if MOE_ASYNC
      __builtin_amdgcn_global_load_async_to_lds_b128(
          (a4i_g)(void*)(bf16_t*)g, (a4i_l)(void*)l, 0, 0);
#else
      *(v8bf*)l = *(const v8bf*)g;
#endif
    }
#if MOE_ASYNC
    __builtin_amdgcn_s_wait_asynccnt(0);
#endif
    __syncthreads();
    // A fragment: lanes 0-15: K k0+0..7 then k0+16..23 ; lanes 16-31: +8
    v16bf af = frag_pair(arow + k0 + half * 8, arow + k0 + 16 + half * 8);
    const bf16_t* lb0 = lds + lr * LDS_PITCH + half * 8;
#pragma unroll
    for (int t = 0; t < 4; ++t) {
      const bf16_t* lb = lb0 + t * 16 * LDS_PITCH;
      v16bf bfm = frag_pair(lb, lb + 16);
      acc[t] = __builtin_amdgcn_wmma_f32_16x16x32_bf16(false, af, false, bfm,
                                                       (short)0, acc[t], false, false);
    }
  }
}

// ---------------- elementwise / setup kernels ----------------

__global__ void k_cvt_bf16(const float* __restrict__ in, bf16_t* __restrict__ out, long n)
{
  long i = (long)blockIdx.x * 256 + threadIdx.x;
  if (i < n) out[i] = (bf16_t)in[i];
}

// w1[i,e,d,h] (D,HID) -> w1T[e,h,d] (HID,D)  : B[N,K] layout for y @ w1
__global__ void k_w1_transpose(const float* __restrict__ w1L, bf16_t* __restrict__ out)
{
  long i = (long)blockIdx.x * 256 + threadIdx.x;
  if (i >= (long)EE * HIDD * DD) return;
  int e   = (int)(i / ((long)HIDD * DD));
  long r  = i % ((long)HIDD * DD);
  int h   = (int)(r / DD);
  int d   = (int)(r % DD);
  out[i] = (bf16_t)w1L[((long)e * DD + d) * HIDD + h];
}

// w2[i,e,h,d] (HID,D) -> w2T[e,d,h] (D,HID)
__global__ void k_w2_transpose(const float* __restrict__ w2L, bf16_t* __restrict__ out)
{
  long i = (long)blockIdx.x * 256 + threadIdx.x;
  if (i >= (long)EE * DD * HIDD) return;
  int e   = (int)(i / ((long)DD * HIDD));
  long r  = i % ((long)DD * HIDD);
  int d   = (int)(r / HIDD);
  int h   = (int)(r % HIDD);
  out[i] = (bf16_t)w2L[((long)e * HIDD + h) * DD + d];
}

__global__ void k_im2col(const float* __restrict__ x, bf16_t* __restrict__ xcol)
{
  long i = (long)blockIdx.x * 256 + threadIdx.x;
  if (i >= (long)MPAT * DD) return;
  int r  = (int)(i / DD), kk = (int)(i % DD);
  int b  = r / NPATCH, pp = r % NPATCH;
  int py = pp / 14, px = pp % 14;
  int c  = kk >> 8, rem = kk & 255, iy = rem >> 4, ix = rem & 15;
  xcol[i] = (bf16_t)x[((long)(b * 3 + c) * 224 + (py * 16 + iy)) * 224 + (px * 16 + ix)];
}

__global__ void k_zero_h(float* __restrict__ hf, bf16_t* __restrict__ hb)
{
  long i = (long)blockIdx.x * 256 + threadIdx.x;
  if (i < (long)MPB * DD) { hf[i] = 0.0f; hb[i] = (bf16_t)0.0f; }
}

__global__ void k_cls(const float* __restrict__ cls, const float* __restrict__ pos,
                      float* __restrict__ hf, bf16_t* __restrict__ hb)
{
  int b = blockIdx.x;
  for (int n = threadIdx.x; n < DD; n += 256) {
    float v = cls[n] + pos[n];
    long o = (long)b * SP * DD + n;
    hf[o] = v; hb[o] = (bf16_t)v;
  }
}

// ---------------- GEMM kernels (128 threads = 4 waves) ----------------

__global__ void __launch_bounds__(128)
k_gemm_patch(const bf16_t* __restrict__ xcol, const bf16_t* __restrict__ pw,
             const float* __restrict__ pb, const float* __restrict__ pos,
             float* __restrict__ hf, bf16_t* __restrict__ hb)
{
  __shared__ bf16_t lds[64 * LDS_PITCH];
  int tid = threadIdx.x, w = tid >> 5, lane = tid & 31, half = lane >> 4, lr = lane & 15;
  int m0 = blockIdx.x * 64 + w * 16, n0 = blockIdx.y * 64;
  int ar = m0 + lr; if (ar > MPAT - 1) ar = MPAT - 1;
  v8f acc[4]; acc_zero(acc);
  gemm_core(xcol + (long)ar * DD, pw, DD, n0, DD, lds, acc);
#pragma unroll
  for (int t = 0; t < 4; ++t)
#pragma unroll
    for (int r = 0; r < 8; ++r) {
      int m = m0 + r + half * 8;
      if (m >= MPAT) continue;
      int n = n0 + t * 16 + lr;
      int b = m / NPATCH, pp = m % NPATCH;
      float v = acc[t][r] + pb[n] + pos[(long)(1 + pp) * DD + n];
      long o = (long)(b * SP + 1 + pp) * DD + n;
      hf[o] = v; hb[o] = (bf16_t)v;
    }
}

__global__ void __launch_bounds__(128)
k_gemm_q(const bf16_t* __restrict__ A, const bf16_t* __restrict__ Bw,
         const float* __restrict__ bias, bf16_t* __restrict__ q)
{
  __shared__ bf16_t lds[64 * LDS_PITCH];
  int tid = threadIdx.x, w = tid >> 5, lane = tid & 31, half = lane >> 4, lr = lane & 15;
  int m0 = blockIdx.x * 64 + w * 16, n0 = blockIdx.y * 64;
  v8f acc[4]; acc_zero(acc);
  gemm_core(A + (long)(m0 + lr) * DD, Bw, DD, n0, DD, lds, acc);
#pragma unroll
  for (int t = 0; t < 4; ++t)
#pragma unroll
    for (int r = 0; r < 8; ++r) {
      int m = m0 + r + half * 8, n = n0 + t * 16 + lr;
      int b = m >> 8, s = m & 255, hh = n >> 6, dd = n & 63;
      q[((long)(b * NHH + hh) * SP + s) * HDD + dd] = (bf16_t)(acc[t][r] + bias[n]);
    }
}

__global__ void __launch_bounds__(128)
k_gemm_kv(const bf16_t* __restrict__ A, const bf16_t* __restrict__ Bw,
          const float* __restrict__ bias, bf16_t* __restrict__ kout,
          bf16_t* __restrict__ vT)
{
  __shared__ bf16_t lds[64 * LDS_PITCH];
  int tid = threadIdx.x, w = tid >> 5, lane = tid & 31, half = lane >> 4, lr = lane & 15;
  int m0 = blockIdx.x * 64 + w * 16, n0 = blockIdx.y * 64;  // n in 0..1535
  v8f acc[4]; acc_zero(acc);
  gemm_core(A + (long)(m0 + lr) * DD, Bw, DD, n0, DD, lds, acc);
#pragma unroll
  for (int t = 0; t < 4; ++t)
#pragma unroll
    for (int r = 0; r < 8; ++r) {
      int m = m0 + r + half * 8, n = n0 + t * 16 + lr;
      float v = acc[t][r] + bias[n];
      int b = m >> 8, s = m & 255;
      if (n < DD) {
        int hh = n >> 6, dd = n & 63;
        kout[((long)(b * NHH + hh) * SP + s) * HDD + dd] = (bf16_t)v;
      } else {
        int n2 = n - DD, hh = n2 >> 6, dd = n2 & 63;
        vT[((long)(b * NHH + hh) * HDD + dd) * SP + s] = (bf16_t)v;  // store V transposed
      }
    }
}

__global__ void __launch_bounds__(128)
k_gemm_scores(const bf16_t* __restrict__ q, const bf16_t* __restrict__ k,
              float* __restrict__ sc)
{
  __shared__ bf16_t lds[64 * LDS_PITCH];
  int tid = threadIdx.x, w = tid >> 5, lane = tid & 31, half = lane >> 4, lr = lane & 15;
  int m0 = blockIdx.x * 64 + w * 16, n0 = blockIdx.y * 64;
  int bh = blockIdx.z;
  const bf16_t* A  = q + (long)bh * SP * HDD;
  const bf16_t* Bm = k + (long)bh * SP * HDD;
  v8f acc[4]; acc_zero(acc);
  gemm_core(A + (long)(m0 + lr) * HDD, Bm, HDD, n0, HDD, lds, acc);
#pragma unroll
  for (int t = 0; t < 4; ++t)
#pragma unroll
    for (int r = 0; r < 8; ++r) {
      int m = m0 + r + half * 8, n = n0 + t * 16 + lr;
      sc[((long)bh * SP + m) * SP + n] = acc[t][r];
    }
}

__global__ void __launch_bounds__(128)
k_gemm_attnv(const bf16_t* __restrict__ at, const bf16_t* __restrict__ vT,
             bf16_t* __restrict__ ob)
{
  __shared__ bf16_t lds[64 * LDS_PITCH];
  int tid = threadIdx.x, w = tid >> 5, lane = tid & 31, half = lane >> 4, lr = lane & 15;
  int m0 = blockIdx.x * 64 + w * 16;
  int bh = blockIdx.z, b = bh / NHH, hh = bh % NHH;
  const bf16_t* A  = at + (long)bh * SP * SP;
  const bf16_t* Bm = vT + (long)bh * HDD * SP;
  v8f acc[4]; acc_zero(acc);
  gemm_core(A + (long)(m0 + lr) * SP, Bm, SP, 0, SP, lds, acc);
#pragma unroll
  for (int t = 0; t < 4; ++t)
#pragma unroll
    for (int r = 0; r < 8; ++r) {
      int m = m0 + r + half * 8, n = t * 16 + lr;
      ob[(long)(b * SP + m) * DD + hh * HDD + n] = (bf16_t)acc[t][r];
    }
}

__global__ void __launch_bounds__(128)
k_gemm_outproj(const bf16_t* __restrict__ ob, const bf16_t* __restrict__ wo,
               const float* __restrict__ bias, float* __restrict__ hf,
               bf16_t* __restrict__ hb)
{
  __shared__ bf16_t lds[64 * LDS_PITCH];
  int tid = threadIdx.x, w = tid >> 5, lane = tid & 31, half = lane >> 4, lr = lane & 15;
  int m0 = blockIdx.x * 64 + w * 16, n0 = blockIdx.y * 64;
  v8f acc[4]; acc_zero(acc);
  gemm_core(ob + (long)(m0 + lr) * DD, wo, DD, n0, DD, lds, acc);
#pragma unroll
  for (int t = 0; t < 4; ++t)
#pragma unroll
    for (int r = 0; r < 8; ++r) {
      int m = m0 + r + half * 8, s = m & 255;
      if (s >= SS) continue;  // keep pad rows zero
      int n = n0 + t * 16 + lr;
      long o = (long)m * DD + n;
      float v = hf[o] + acc[t][r] + bias[n];
      hf[o] = v; hb[o] = (bf16_t)v;
    }
}

__global__ void __launch_bounds__(128)
k_gemm_moe1(const bf16_t* __restrict__ y, const bf16_t* __restrict__ w1T,
            const float* __restrict__ b1e, const int* __restrict__ perm_e,
            const int* __restrict__ cnt_e, bf16_t* __restrict__ hid)
{
  __shared__ bf16_t lds[64 * LDS_PITCH];
  int c = *cnt_e;
  if ((int)blockIdx.x * 64 >= c) return;  // uniform early exit
  int tid = threadIdx.x, w = tid >> 5, lane = tid & 31, half = lane >> 4, lr = lane & 15;
  int m0 = blockIdx.x * 64 + w * 16, n0 = blockIdx.y * 64;
  int idx = m0 + lr; if (idx > c - 1) idx = c - 1;
  v8f acc[4]; acc_zero(acc);
  gemm_core(y + (long)perm_e[idx] * DD, w1T, DD, n0, DD, lds, acc);
#pragma unroll
  for (int t = 0; t < 4; ++t)
#pragma unroll
    for (int r = 0; r < 8; ++r) {
      int m = m0 + r + half * 8;
      if (m >= c) continue;
      int n = n0 + t * 16 + lr;
      hid[(long)m * HIDD + n] = (bf16_t)gelu_tanh(acc[t][r] + b1e[n]);
    }
}

__global__ void __launch_bounds__(128)
k_gemm_moe2(const bf16_t* __restrict__ hid, const bf16_t* __restrict__ w2T,
            const float* __restrict__ b2e, const int* __restrict__ perm_e,
            const int* __restrict__ cnt_e, const float* __restrict__ gv,
            float* __restrict__ hf, bf16_t* __restrict__ hb)
{
  __shared__ bf16_t lds[64 * LDS_PITCH];
  int c = *cnt_e;
  if ((int)blockIdx.x * 64 >= c) return;
  int tid = threadIdx.x, w = tid >> 5, lane = tid & 31, half = lane >> 4, lr = lane & 15;
  int m0 = blockIdx.x * 64 + w * 16, n0 = blockIdx.y * 64;
  int idx = m0 + lr; if (idx > c - 1) idx = c - 1;
  v8f acc[4]; acc_zero(acc);
  gemm_core(hid + (long)idx * HIDD, w2T, HIDD, n0, HIDD, lds, acc);
#pragma unroll
  for (int t = 0; t < 4; ++t)
#pragma unroll
    for (int r = 0; r < 8; ++r) {
      int m = m0 + r + half * 8;
      if (m >= c) continue;
      int n = n0 + t * 16 + lr;
      int tok = perm_e[m];
      long o = (long)tok * DD + n;
      float v = hf[o] + (acc[t][r] + b2e[n]) * gv[tok];
      hf[o] = v; hb[o] = (bf16_t)v;
    }
}

// ---------------- reductions / misc ----------------

__global__ void __launch_bounds__(256)
k_layernorm(const float* __restrict__ h, const float* __restrict__ g,
            const float* __restrict__ b, bf16_t* __restrict__ out)
{
  int row = blockIdx.x;
  const float* x = h + (long)row * DD;
  __shared__ float ss[256], ss2[256];
  float s = 0.f, s2 = 0.f;
  for (int i = threadIdx.x; i < DD; i += 256) { float v = x[i]; s += v; s2 += v * v; }
  ss[threadIdx.x] = s; ss2[threadIdx.x] = s2; __syncthreads();
  for (int st = 128; st; st >>= 1) {
    if ((int)threadIdx.x < st) { ss[threadIdx.x] += ss[threadIdx.x + st]; ss2[threadIdx.x] += ss2[threadIdx.x + st]; }
    __syncthreads();
  }
  float mean = ss[0] * (1.0f / DD);
  float var  = ss2[0] * (1.0f / DD) - mean * mean;
  float rstd = rsqrtf(var + 1e-5f);
  for (int i = threadIdx.x; i < DD; i += 256)
    out[(long)row * DD + i] = (bf16_t)((x[i] - mean) * rstd * g[i] + b[i]);
}

__global__ void __launch_bounds__(256)
k_softmax(const float* __restrict__ sc, bf16_t* __restrict__ at)
{
  long row = blockIdx.x;            // 0 .. B*NH*SP-1
  int sq = (int)(row & (SP - 1));
  int t = threadIdx.x;
  const float* sr = sc + row * SP;
  bf16_t* ar = at + row * SP;
  if (sq >= SS) { ar[t] = (bf16_t)0.0f; return; }  // pad q rows -> zero attn
  __shared__ float red[256];
  float v = (t < SS) ? sr[t] * 0.125f : -3.0e38f;  // scale = 1/sqrt(64)
  red[t] = v; __syncthreads();
  for (int st = 128; st; st >>= 1) {
    if (t < st) red[t] = fmaxf(red[t], red[t + st]);
    __syncthreads();
  }
  float mx = red[0]; __syncthreads();
  float e = (t < SS) ? __expf(v - mx) : 0.0f;
  red[t] = e; __syncthreads();
  for (int st = 128; st; st >>= 1) {
    if (t < st) red[t] += red[t + st];
    __syncthreads();
  }
  ar[t] = (bf16_t)(e / red[0]);
}

__global__ void k_zero_cnt(int* __restrict__ cnt)
{
  if (threadIdx.x < EE) cnt[threadIdx.x] = 0;
}

__global__ void __launch_bounds__(256)
k_gate(const bf16_t* __restrict__ y, const float* __restrict__ gw,
       const float* __restrict__ gb, int* __restrict__ perm,
       int* __restrict__ cnt, float* __restrict__ gv)
{
  int m = blockIdx.x;
  int s = m & (SP - 1);
  if (s >= SS) return;              // uniform per block
  __shared__ float red[256];
  int t = threadIdx.x, e = t >> 6, j = t & 63;
  const bf16_t* yr = y + (long)m * DD;
  const float*  wr = gw + (long)e * DD;
  float a = 0.f;
  for (int i = j; i < DD; i += 64) a += (float)yr[i] * wr[i];
  red[t] = a; __syncthreads();
  for (int st = 32; st; st >>= 1) {
    if (j < st) red[t] += red[t + st];
    __syncthreads();
  }
  if (t == 0) {
    float l[EE];
#pragma unroll
    for (int q = 0; q < EE; ++q) l[q] = red[q * 64] + gb[q];
    float mx = l[0]; int top = 0;
#pragma unroll
    for (int q = 1; q < EE; ++q) if (l[q] > mx) { mx = l[q]; top = q; }  // first-max, as argmax
    float sum = 0.f;
#pragma unroll
    for (int q = 0; q < EE; ++q) sum += __expf(l[q] - mx);
    gv[m] = 1.0f / sum;             // softmax prob of the top expert
    int slot = atomicAdd(&cnt[top], 1);
    perm[top * MPB + slot] = m;
  }
}

__global__ void __launch_bounds__(256)
k_head(const float* __restrict__ hf, const float* __restrict__ hw,
       const float* __restrict__ hb2, float* __restrict__ out)
{
  int b = blockIdx.x;
  int n = blockIdx.y * 256 + threadIdx.x;
  if (n >= NCC) return;
  const float* xr = hf + (long)b * SP * DD;     // cls row (s = 0)
  const float* wr = hw + (long)n * DD;
  float a = 0.f;
  for (int i = 0; i < DD; ++i) a += xr[i] * wr[i];
  out[b * NCC + n] = a + hb2[n];
}

// ---------------- host launch ----------------

extern "C" void kernel_launch(void* const* d_in, const int* in_sizes, int n_in,
                              void* d_out, int out_size, void* d_ws, size_t ws_size,
                              hipStream_t stream)
{
  (void)in_sizes; (void)n_in; (void)out_size; (void)ws_size;
  const float* x          = (const float*)d_in[0];
  const float* patch_w    = (const float*)d_in[1];
  const float* patch_b    = (const float*)d_in[2];
  const float* cls_tok    = (const float*)d_in[3];
  const float* pos_emb    = (const float*)d_in[4];
  const float* ln1_g      = (const float*)d_in[5];
  const float* ln1_b      = (const float*)d_in[6];
  const float* ln2_g      = (const float*)d_in[7];
  const float* ln2_b      = (const float*)d_in[8];
  const float* attn_in_w  = (const float*)d_in[9];
  const float* attn_in_b  = (const float*)d_in[10];
  const float* attn_out_w = (const float*)d_in[11];
  const float* attn_out_b = (const float*)d_in[12];
  const float* gate_w     = (const float*)d_in[13];
  const float* gate_b     = (const float*)d_in[14];
  const float* w1         = (const float*)d_in[15];
  const float* b1         = (const float*)d_in[16];
  const float* w2         = (const float*)d_in[17];
  const float* b2         = (const float*)d_in[18];
  const float* head_w     = (const float*)d_in[19];
  const float* head_b     = (const float*)d_in[20];
  float* out = (float*)d_out;

  char* wsp = (char*)d_ws;
  auto take = [&wsp](size_t bytes) -> void* {
    void* r = (void*)wsp;
    wsp += (bytes + 255) & ~(size_t)255;
    return r;
  };
  float*  hf   = (float*) take((size_t)MPB * DD * 4);
  bf16_t* hb   = (bf16_t*)take((size_t)MPB * DD * 2);
  bf16_t* qn   = (bf16_t*)take((size_t)MPB * DD * 2);
  bf16_t* yb   = (bf16_t*)take((size_t)MPB * DD * 2);
  bf16_t* ob   = (bf16_t*)take((size_t)MPB * DD * 2);
  bf16_t* qB   = (bf16_t*)take((size_t)BB * NHH * SP * HDD * 2);
  bf16_t* kB   = (bf16_t*)take((size_t)BB * NHH * SP * HDD * 2);
  bf16_t* vT   = (bf16_t*)take((size_t)BB * NHH * HDD * SP * 2);
  float*  sc   = (float*) take((size_t)BB * NHH * SP * SP * 4);
  bf16_t* at   = (bf16_t*)take((size_t)BB * NHH * SP * SP * 2);
  bf16_t* hidb = (bf16_t*)take((size_t)MPB * HIDD * 2);
  bf16_t* xcol = (bf16_t*)take((size_t)MPAT * DD * 2);
  bf16_t* pwb  = (bf16_t*)take((size_t)DD * DD * 2);
  bf16_t* wqkv = (bf16_t*)take((size_t)3 * DD * DD * 2);
  bf16_t* wo   = (bf16_t*)take((size_t)DD * DD * 2);
  bf16_t* w1T  = (bf16_t*)take((size_t)EE * HIDD * DD * 2);
  bf16_t* w2T  = (bf16_t*)take((size_t)EE * DD * HIDD * 2);
  float*  gv   = (float*) take((size_t)MPB * 4);
  int*    cnt  = (int*)   take((size_t)EE * 4);
  int*    perm = (int*)   take((size_t)EE * MPB * 4);

  const dim3 blk128(128), blk256(256);
  auto blocks = [](long n) { return (unsigned)((n + 255) / 256); };

  // ---- patch embedding ----
  k_cvt_bf16<<<blocks((long)DD * DD), blk256, 0, stream>>>(patch_w, pwb, (long)DD * DD);
  k_im2col  <<<blocks((long)MPAT * DD), blk256, 0, stream>>>(x, xcol);
  k_zero_h  <<<blocks((long)MPB * DD), blk256, 0, stream>>>(hf, hb);
  k_gemm_patch<<<dim3((MPAT + 63) / 64, DD / 64), blk128, 0, stream>>>(xcol, pwb, patch_b, pos_emb, hf, hb);
  k_cls<<<dim3(BB), blk256, 0, stream>>>(cls_tok, pos_emb, hf, hb);

  // ---- transformer layers ----
  for (int i = 0; i < DEPTHH; ++i) {
    k_cvt_bf16<<<blocks((long)3 * DD * DD), blk256, 0, stream>>>(attn_in_w + (long)i * 3 * DD * DD, wqkv, (long)3 * DD * DD);
    k_cvt_bf16<<<blocks((long)DD * DD), blk256, 0, stream>>>(attn_out_w + (long)i * DD * DD, wo, (long)DD * DD);
    k_w1_transpose<<<blocks((long)EE * HIDD * DD), blk256, 0, stream>>>(w1 + (long)i * EE * DD * HIDD, w1T);
    k_w2_transpose<<<blocks((long)EE * DD * HIDD), blk256, 0, stream>>>(w2 + (long)i * EE * HIDD * DD, w2T);

    k_layernorm<<<dim3(MPB), blk256, 0, stream>>>(hf, ln1_g + (long)i * DD, ln1_b + (long)i * DD, qn);

    // q from LN(h); k,v from raw h (per reference)
    k_gemm_q <<<dim3(MPB / 64, DD / 64), blk128, 0, stream>>>(qn, wqkv, attn_in_b + (long)i * 3 * DD, qB);
    k_gemm_kv<<<dim3(MPB / 64, 2 * DD / 64), blk128, 0, stream>>>(hb, wqkv + (long)DD * DD,
                                                                  attn_in_b + (long)i * 3 * DD + DD, kB, vT);

    k_gemm_scores<<<dim3(SP / 64, SP / 64, BB * NHH), blk128, 0, stream>>>(qB, kB, sc);
    k_softmax<<<dim3(BB * NHH * SP), blk256, 0, stream>>>(sc, at);
    k_gemm_attnv<<<dim3(SP / 64, 1, BB * NHH), blk128, 0, stream>>>(at, vT, ob);
    k_gemm_outproj<<<dim3(MPB / 64, DD / 64), blk128, 0, stream>>>(ob, wo, attn_out_b + (long)i * DD, hf, hb);

    k_layernorm<<<dim3(MPB), blk256, 0, stream>>>(hf, ln2_g + (long)i * DD, ln2_b + (long)i * DD, yb);

    k_zero_cnt<<<dim3(1), dim3(32), 0, stream>>>(cnt);
    k_gate<<<dim3(MPB), blk256, 0, stream>>>(yb, gate_w + (long)i * EE * DD, gate_b + (long)i * EE, perm, cnt, gv);

    for (int e = 0; e < EE; ++e) {
      k_gemm_moe1<<<dim3(MPB / 64, HIDD / 64), blk128, 0, stream>>>(
          yb, w1T + (long)e * HIDD * DD, b1 + (long)(i * EE + e) * HIDD,
          perm + (long)e * MPB, cnt + e, hidb);
      k_gemm_moe2<<<dim3(MPB / 64, DD / 64), blk128, 0, stream>>>(
          hidb, w2T + (long)e * DD * HIDD, b2 + (long)(i * EE + e) * DD,
          perm + (long)e * MPB, cnt + e, gv, hf, hb);
    }
  }

  // ---- classifier head ----
  k_head<<<dim3(BB, (NCC + 255) / 256), blk256, 0, stream>>>(hf, head_w, head_b, out);
}
